// GCN_72645076844749
// MI455X (gfx1250) — compile-verified, matrix operand order checked by reading, not run on hardware
//
#include <hip/hip_runtime.h>

typedef __attribute__((ext_vector_type(16))) _Float16 v16h;
typedef __attribute__((ext_vector_type(8)))  _Float16 v8h;
typedef __attribute__((ext_vector_type(8)))  float    v8f;
typedef __attribute__((ext_vector_type(4)))  float    v4f;

#define GCN_N    10000
#define GCN_KPAD 10016   // K padded to multiple of 32 for zero-padded f16 buffers
#define GCN_F    512
#define GCN_H    256
#define GCN_C    40
#define GCN_CP   48      // classes padded to 3x16 tiles

__device__ __forceinline__ v8f wmma_f16(v16h a, v16h b, v8f c) {
  // D = A(16x32 f16) x B(32x16 f16) + C(16x16 f32)
  return __builtin_amdgcn_wmma_f32_16x16x32_f16(false, a, false, b, (short)0, c, false, false);
}

// 16x32 f16 fragment from a row-major f16 matrix. Lane holds row (A) / col (B) = lane&15.
// ISA layout: VGPRs 0-3 hold K = 8*half + 0..7, VGPRs 4-7 hold K = 16 + 8*half + 0..7.
__device__ __forceinline__ v16h frag_f16(const _Float16* __restrict__ src, int ld,
                                         int row0, int rowMax, int k0) {
  int l  = threadIdx.x & 31;
  int r  = row0 + (l & 15);
  r = r > rowMax ? rowMax : r;
  int h8 = (l >> 4) << 3;
  const _Float16* p = src + (size_t)r * ld + k0 + h8;
  v8h lo = *(const v8h*)(p);
  v8h hi = *(const v8h*)(p + 16);
  return __builtin_shufflevector(lo, hi, 0,1,2,3,4,5,6,7,8,9,10,11,12,13,14,15);
}

// 16x32 fragment from row-major f32 matrix with on-the-fly f16 convert.
// If !FULL, only K 0..15 are valid (K tail of 16); upper half is zeroed.
template<bool FULL>
__device__ __forceinline__ v16h frag_f32(const float* __restrict__ src, int ld,
                                         int row0, int rowMax, int k0) {
  int l  = threadIdx.x & 31;
  int r  = row0 + (l & 15);
  r = r > rowMax ? rowMax : r;
  int h8 = (l >> 4) << 3;
  const float* p = src + (size_t)r * ld + k0 + h8;
  v4f a0 = *(const v4f*)(p);
  v4f a1 = *(const v4f*)(p + 4);
  v16h out;
  #pragma unroll
  for (int i = 0; i < 4; ++i) { out[i] = (_Float16)a0[i]; out[4+i] = (_Float16)a1[i]; }
  if (FULL) {
    v4f b0 = *(const v4f*)(p + 16);
    v4f b1 = *(const v4f*)(p + 20);
    #pragma unroll
    for (int i = 0; i < 4; ++i) { out[8+i] = (_Float16)b0[i]; out[12+i] = (_Float16)b1[i]; }
  } else {
    #pragma unroll
    for (int i = 0; i < 8; ++i) out[8+i] = (_Float16)0.0f;
  }
  return out;
}

// ---------------- prep kernels ----------------
__global__ void k_zero32(unsigned int* __restrict__ p, int n) {
  int i = blockIdx.x * blockDim.x + threadIdx.x;
  if (i < n) p[i] = 0u;
}
__global__ void k_w1t(const float* __restrict__ w1, _Float16* __restrict__ w1t) {
  int idx = blockIdx.x * blockDim.x + threadIdx.x;   // over 256*512
  if (idx >= GCN_H * GCN_F) return;
  int h = idx / GCN_F, k = idx % GCN_F;
  w1t[idx] = (_Float16)w1[(size_t)k * GCN_H + h];
}
__global__ void k_w2t(const float* __restrict__ w2, _Float16* __restrict__ w2t) {
  int idx = blockIdx.x * blockDim.x + threadIdx.x;   // over 48*256
  if (idx >= GCN_CP * GCN_H) return;
  int c = idx / GCN_H, k = idx % GCN_H;
  w2t[idx] = (c < GCN_C) ? (_Float16)w2[(size_t)k * GCN_C + c] : (_Float16)0.0f;
}

// ---------------- GEMM 1: XWT[h][n] = sum_k feature[n][k] * W1T[h][k] ----------------
// D[n,h] tiles; stored transposed (n contiguous) as packed b128 f16 stores.
__global__ __launch_bounds__(128) void k_gemm1(const float* __restrict__ feat,
                                               const _Float16* __restrict__ w1t,
                                               _Float16* __restrict__ xwt) {
  int wave  = threadIdx.x >> 5;
  int nbase = (blockIdx.y * 2 + wave) * 64;
  int hbase = blockIdx.x * 64;
  int l = threadIdx.x & 31;
  v8f acc[4][4] = {};
  for (int kk = 0; kk < GCN_F; kk += 32) {
    v16h a[4], b[4];
    #pragma unroll
    for (int i = 0; i < 4; ++i) a[i] = frag_f32<true>(feat, GCN_F, nbase + 16*i, GCN_N-1, kk);
    #pragma unroll
    for (int j = 0; j < 4; ++j) b[j] = frag_f16(w1t, GCN_F, hbase + 16*j, GCN_H-1, kk);
    #pragma unroll
    for (int i = 0; i < 4; ++i)
      #pragma unroll
      for (int j = 0; j < 4; ++j) acc[i][j] = wmma_f16(a[i], b[j], acc[i][j]);
  }
  int half8 = (l >> 4) << 3;
  #pragma unroll
  for (int i = 0; i < 4; ++i) {
    if (nbase + 16*i >= GCN_N) continue;          // N % 16 == 0 -> whole-tile guard
    int n0 = nbase + 16*i + half8;
    #pragma unroll
    for (int j = 0; j < 4; ++j) {
      int h = hbase + 16*j + (l & 15);
      v8h o;
      #pragma unroll
      for (int r = 0; r < 8; ++r) o[r] = (_Float16)acc[i][j][r];
      *(v8h*)(xwt + (size_t)h * GCN_KPAD + n0) = o;
    }
  }
}

// ---------------- GEMM 2: H[m][h] = relu(adj[m][:] @ XW[:,h] + b1[h]) ----------------
__global__ __launch_bounds__(128) void k_gemm2(const float* __restrict__ adj,
                                               const _Float16* __restrict__ xwt,
                                               const float* __restrict__ b1,
                                               _Float16* __restrict__ Hm) {
  int wave  = threadIdx.x >> 5;
  int mbase = (blockIdx.y * 2 + wave) * 64;
  int hbase = blockIdx.x * 64;
  int l = threadIdx.x & 31;
  v8f acc[4][4] = {};
  int kk = 0;
  for (; kk + 32 <= GCN_N; kk += 32) {
    v16h a[4], b[4];
    #pragma unroll
    for (int i = 0; i < 4; ++i) a[i] = frag_f32<true>(adj, GCN_N, mbase + 16*i, GCN_N-1, kk);
    #pragma unroll
    for (int j = 0; j < 4; ++j) b[j] = frag_f16(xwt, GCN_KPAD, hbase + 16*j, GCN_H-1, kk);
    #pragma unroll
    for (int i = 0; i < 4; ++i)
      #pragma unroll
      for (int j = 0; j < 4; ++j) acc[i][j] = wmma_f16(a[i], b[j], acc[i][j]);
  }
  { // K tail of 16 (10000 % 32); XWT is zero-padded so B is a normal load
    v16h a[4], b[4];
    #pragma unroll
    for (int i = 0; i < 4; ++i) a[i] = frag_f32<false>(adj, GCN_N, mbase + 16*i, GCN_N-1, kk);
    #pragma unroll
    for (int j = 0; j < 4; ++j) b[j] = frag_f16(xwt, GCN_KPAD, hbase + 16*j, GCN_H-1, kk);
    #pragma unroll
    for (int i = 0; i < 4; ++i)
      #pragma unroll
      for (int j = 0; j < 4; ++j) acc[i][j] = wmma_f16(a[i], b[j], acc[i][j]);
  }
  float bj[4];
  #pragma unroll
  for (int j = 0; j < 4; ++j) bj[j] = b1[hbase + 16*j + (l & 15)];
  int half8 = (l >> 4) << 3;
  #pragma unroll
  for (int i = 0; i < 4; ++i) {
    if (mbase + 16*i >= GCN_N) continue;
    #pragma unroll
    for (int j = 0; j < 4; ++j) {
      int h = hbase + 16*j + (l & 15);
      #pragma unroll
      for (int r = 0; r < 8; ++r) {
        float v = acc[i][j][r] + bj[j];
        v = v > 0.0f ? v : 0.0f;
        Hm[(size_t)(mbase + 16*i + half8 + r) * GCN_H + h] = (_Float16)v;
      }
    }
  }
}

// ---------------- GEMM 3: HWT[c][n] = sum_h H[n][h] * W2T[c][h] ----------------
__global__ __launch_bounds__(128) void k_gemm3(const _Float16* __restrict__ Hm,
                                               const _Float16* __restrict__ w2t,
                                               _Float16* __restrict__ hwt) {
  int wave  = threadIdx.x >> 5;
  int nbase = (blockIdx.x * 2 + wave) * 64;
  int l = threadIdx.x & 31;
  v8f acc[4][3] = {};
  for (int kk = 0; kk < GCN_H; kk += 32) {
    v16h a[4], b[3];
    #pragma unroll
    for (int i = 0; i < 4; ++i) a[i] = frag_f16(Hm, GCN_H, nbase + 16*i, GCN_N-1, kk);
    #pragma unroll
    for (int j = 0; j < 3; ++j) b[j] = frag_f16(w2t, GCN_H, 16*j, GCN_CP-1, kk);
    #pragma unroll
    for (int i = 0; i < 4; ++i)
      #pragma unroll
      for (int j = 0; j < 3; ++j) acc[i][j] = wmma_f16(a[i], b[j], acc[i][j]);
  }
  int half8 = (l >> 4) << 3;
  #pragma unroll
  for (int i = 0; i < 4; ++i) {
    if (nbase + 16*i >= GCN_N) continue;
    int n0 = nbase + 16*i + half8;
    #pragma unroll
    for (int j = 0; j < 3; ++j) {
      int c = 16*j + (l & 15);
      v8h o;
      #pragma unroll
      for (int r = 0; r < 8; ++r) o[r] = (_Float16)acc[i][j][r];
      *(v8h*)(hwt + (size_t)c * GCN_KPAD + n0) = o;
    }
  }
}

// ------- GEMM 4 + fused log_softmax: out[m][c] = logsoftmax(adj@HW + b2) -------
__global__ __launch_bounds__(128) void k_gemm4(const float* __restrict__ adj,
                                               const _Float16* __restrict__ hwt,
                                               const float* __restrict__ b2,
                                               float* __restrict__ out) {
  int wave  = threadIdx.x >> 5;
  int mbase = (blockIdx.x * 2 + wave) * 64;
  int l = threadIdx.x & 31;
  v8f acc[4][3] = {};
  int kk = 0;
  for (; kk + 32 <= GCN_N; kk += 32) {
    v16h a[4], b[3];
    #pragma unroll
    for (int i = 0; i < 4; ++i) a[i] = frag_f32<true>(adj, GCN_N, mbase + 16*i, GCN_N-1, kk);
    #pragma unroll
    for (int j = 0; j < 3; ++j) b[j] = frag_f16(hwt, GCN_KPAD, 16*j, GCN_CP-1, kk);
    #pragma unroll
    for (int i = 0; i < 4; ++i)
      #pragma unroll
      for (int j = 0; j < 3; ++j) acc[i][j] = wmma_f16(a[i], b[j], acc[i][j]);
  }
  { // K tail of 16
    v16h a[4], b[3];
    #pragma unroll
    for (int i = 0; i < 4; ++i) a[i] = frag_f32<false>(adj, GCN_N, mbase + 16*i, GCN_N-1, kk);
    #pragma unroll
    for (int j = 0; j < 3; ++j) b[j] = frag_f16(hwt, GCN_KPAD, 16*j, GCN_CP-1, kk);
    #pragma unroll
    for (int i = 0; i < 4; ++i)
      #pragma unroll
      for (int j = 0; j < 3; ++j) acc[i][j] = wmma_f16(a[i], b[j], acc[i][j]);
  }
  int cl = l & 15;
  float bj[3];
  #pragma unroll
  for (int j = 0; j < 3; ++j) { int c = 16*j + cl; bj[j] = (c < GCN_C) ? b2[c] : 0.0f; }
  bool v2 = cl < 8;                 // third tile valid only for c = 32..39
  int half8 = (l >> 4) << 3;
  #pragma unroll
  for (int i = 0; i < 4; ++i) {
    if (mbase + 16*i >= GCN_N) continue;
    #pragma unroll
    for (int r = 0; r < 8; ++r) {
      float x0 = acc[i][0][r] + bj[0];
      float x1 = acc[i][1][r] + bj[1];
      float x2 = acc[i][2][r] + bj[2];
      float mx = fmaxf(x0, x1);
      if (v2) mx = fmaxf(mx, x2);
      #pragma unroll
      for (int off = 1; off < 16; off <<= 1) mx = fmaxf(mx, __shfl_xor(mx, off, 32));
      float s = __expf(x0 - mx) + __expf(x1 - mx) + (v2 ? __expf(x2 - mx) : 0.0f);
      #pragma unroll
      for (int off = 1; off < 16; off <<= 1) s += __shfl_xor(s, off, 32);
      float lse = mx + __logf(s);
      int m = mbase + 16*i + half8 + r;
      float* po = out + (size_t)m * GCN_C;
      po[cl]      = x0 - lse;
      po[16 + cl] = x1 - lse;
      if (v2) po[32 + cl] = x2 - lse;
    }
  }
}

extern "C" void kernel_launch(void* const* d_in, const int* in_sizes, int n_in,
                              void* d_out, int out_size, void* d_ws, size_t ws_size,
                              hipStream_t stream) {
  (void)in_sizes; (void)n_in; (void)out_size; (void)ws_size;
  const float* feat = (const float*)d_in[0];
  const float* adj  = (const float*)d_in[1];
  const float* W1   = (const float*)d_in[2];
  const float* b1   = (const float*)d_in[3];
  const float* W2   = (const float*)d_in[4];
  const float* b2   = (const float*)d_in[5];
  float* out = (float*)d_out;

  char* ws = (char*)d_ws;
  size_t off = 0;
  auto take = [&](size_t bytes) -> char* {
    char* p = ws + off;
    off = (off + bytes + 255) & ~(size_t)255;
    return p;
  };
  _Float16* w1t = (_Float16*)take((size_t)GCN_H * GCN_F * 2);       //  256 KB
  _Float16* w2t = (_Float16*)take((size_t)GCN_CP * GCN_H * 2);      //   24 KB
  _Float16* xwt = (_Float16*)take((size_t)GCN_H * GCN_KPAD * 2);    // ~4.9 MB
  _Float16* Hm  = (_Float16*)take((size_t)GCN_N * GCN_H * 2);       // ~4.9 MB
  _Float16* hwt = (_Float16*)take((size_t)GCN_CP * GCN_KPAD * 2);   // ~0.9 MB

  // zero-fill K-padded f16 buffers (as u32 words)
  {
    int n = GCN_H * GCN_KPAD / 2;
    k_zero32<<<(n + 255) / 256, 256, 0, stream>>>((unsigned int*)xwt, n);
  }
  {
    int n = GCN_CP * GCN_KPAD / 2;
    k_zero32<<<(n + 255) / 256, 256, 0, stream>>>((unsigned int*)hwt, n);
  }
  k_w1t<<<(GCN_H * GCN_F + 255) / 256, 256, 0, stream>>>(W1, w1t);
  k_w2t<<<(GCN_CP * GCN_H + 255) / 256, 256, 0, stream>>>(W2, w2t);

  dim3 g1(4, 79);  // 4 col-groups of 64 over H=256; 79 * 128 rows >= 10000
  k_gemm1<<<g1, 128, 0, stream>>>(feat, w1t, xwt);
  dim3 g2(4, 79);
  k_gemm2<<<g2, 128, 0, stream>>>(adj, xwt, b1, Hm);
  k_gemm3<<<79, 128, 0, stream>>>(Hm, w2t, hwt);
  k_gemm4<<<79, 128, 0, stream>>>(adj, hwt, b2, out);
}
